// PNN_30717606101543
// MI455X (gfx1250) — compile-verified, matrix operand order checked by reading, not run on hardware
//
#include <hip/hip_runtime.h>
#include <hip/hip_bf16.h>
#include <stdint.h>

typedef unsigned short u16;
typedef unsigned int   u32;
typedef unsigned long long u64;

typedef __attribute__((ext_vector_type(16))) __bf16 v16bf;
typedef __attribute__((ext_vector_type(8)))  float  v8f;
typedef __attribute__((ext_vector_type(8)))  u16    u16x8;
typedef __attribute__((ext_vector_type(4)))  u32    u32x4;
typedef __attribute__((ext_vector_type(8)))  int    i32x8;
typedef __attribute__((ext_vector_type(4)))  int    i32x4;

#define N_ROWS 4096
#define M_ROWS 8192
#define D_DIM  512

#define BN 128
#define BM 128
#define BK 64
#define LDSPITCH 72   /* BK + 8 bf16 of pad -> 144-byte row pitch (TDM pad: 16B per 128B) */
#define NCHUNK (D_DIM / BK)

#if defined(__has_builtin)
#if __has_builtin(__builtin_amdgcn_tensor_load_to_lds)
#define HAVE_TDM 1
#endif
#if __has_builtin(__builtin_amdgcn_s_wait_tensorcnt)
#define HAVE_WAIT_TENSOR 1
#endif
#endif

// ---------- small helpers ----------

__device__ __forceinline__ u16 f2bf(float f) {
  u32 u = __float_as_uint(f);
  u32 r = u + 0x7FFFu + ((u >> 16) & 1u);   // round-to-nearest-even
  return (u16)(r >> 16);
}

#ifdef HAVE_TDM
// 2D TDM load: tile_dim0 = BK bf16 elems, tile_dim1 = 128 rows, data_size = 2B,
// pad_enable with interval code 4 (32 DWORDs = 128B) and amount code 3 (4 DWORDs = 16B)
// => LDS row pitch = 144B = LDSPITCH elements.
__device__ __forceinline__ void tdm_load_tile(u32 lds_off, const u16* gptr,
                                              u32 tensor_d1) {
  u64 ga = (u64)(uintptr_t)gptr;
  u32x4 g0;
  g0[0] = 1u;                                   // count=1, is_restore=0, gather off
  g0[1] = lds_off;                              // lds_addr (bytes)
  g0[2] = (u32)ga;                              // global_addr[31:0]
  g0[3] = (u32)((ga >> 32) & 0x1FFFFFFu) | (2u << 30);  // addr[56:32] | type=2

  const u32 tensor_d0 = (u32)D_DIM;
  const u32 tile_d0   = (u32)BK;
  const u32 tile_d1   = 128u;
  const u32 stride0   = (u32)D_DIM;

  i32x8 g1;
  g1[0] = (int)((1u << 16)        // data_size = 2 bytes
              | (1u << 20)        // pad_enable
              | (4u << 22)        // pad_interval code 4 -> every 32 DWORDs (128B)
              | (3u << 25));      // pad_amount code 3  -> 4 DWORDs (16B)
  g1[1] = (int)((tensor_d0 & 0xFFFFu) << 16);              // tensor_dim0 lo16 @ [63:48]
  g1[2] = (int)((tensor_d0 >> 16) | ((tensor_d1 & 0xFFFFu) << 16));
  g1[3] = (int)((tensor_d1 >> 16) | (tile_d0 << 16));      // tile_dim0 @ [127:112]
  g1[4] = (int)tile_d1;                                    // tile_dim1; tile_dim2=0
  g1[5] = (int)stride0;                                    // tensor_dim0_stride lo32
  g1[6] = 0;
  g1[7] = 0;

  i32x4 z4 = {0, 0, 0, 0};
#if __clang_major__ >= 23
  i32x8 z8 = {0, 0, 0, 0, 0, 0, 0, 0};
  __builtin_amdgcn_tensor_load_to_lds(g0, g1, z4, z4, z8, 0);
#else
  __builtin_amdgcn_tensor_load_to_lds(g0, g1, z4, z4, 0);
#endif
}
#endif

__device__ __forceinline__ void wait_tensor0() {
#ifdef HAVE_WAIT_TENSOR
  __builtin_amdgcn_s_wait_tensorcnt(0);
#else
  asm volatile("s_wait_tensorcnt 0x0" ::: "memory");
#endif
}

// ---------- prep kernels ----------

__global__ void pnn_zero(float* __restrict__ out, int n) {
  int i = blockIdx.x * blockDim.x + threadIdx.x;
  if (i < n) out[i] = 0.0f;
}

__global__ void pnn_weights(const int* __restrict__ targets,
                            float* __restrict__ w0, float* __restrict__ w1, int m) {
  __shared__ int cnt[2];
  int tid = threadIdx.x;
  if (tid < 2) cnt[tid] = 0;
  __syncthreads();
  int c0 = 0, c1 = 0;
  for (int i = tid; i < m; i += blockDim.x) {
    int t = targets[i];
    c0 += (t == 0);
    c1 += (t == 1);
  }
  atomicAdd(&cnt[0], c0);
  atomicAdd(&cnt[1], c1);
  __syncthreads();
  float inv0 = 1.0f / fmaxf((float)cnt[0], 1.0f);
  float inv1 = 1.0f / fmaxf((float)cnt[1], 1.0f);
  for (int i = tid; i < m; i += blockDim.x) {
    int t = targets[i];
    w0[i] = (t == 0) ? inv0 : 0.0f;
    w1[i] = (t == 1) ? inv1 : 0.0f;
  }
}

// convert f32 rows -> bf16 rows, exact f32 row squared-norms. One wave per row.
__global__ __launch_bounds__(256)
void pnn_convert(const float* __restrict__ src, u16* __restrict__ dst,
                 float* __restrict__ norms) {
  int lane = threadIdx.x & 31;
  int wave = threadIdx.x >> 5;
  int row  = blockIdx.x * 8 + wave;
  const float* r = src + (size_t)row * D_DIM;
  u16* w = dst + (size_t)row * D_DIM;
  float ss = 0.0f;
  #pragma unroll
  for (int d = lane; d < D_DIM; d += 32) {
    float v = r[d];
    ss += v * v;
    w[d] = f2bf(v);
  }
  #pragma unroll
  for (int o = 16; o > 0; o >>= 1) ss += __shfl_xor(ss, o, 32);
  if (lane == 0) norms[row] = ss;
}

// ---------- main fused GEMM + RBF + masked reduce (double-buffered TDM) ----------

__global__ __launch_bounds__(256)
void pnn_gemm(const u16* __restrict__ xb, const u16* __restrict__ pb,
              const float* __restrict__ x2, const float* __restrict__ p2,
              const float* __restrict__ w0, const float* __restrict__ w1,
              float* __restrict__ out) {
  __shared__ __align__(16) u16 Xs[2][BN * LDSPITCH];
  __shared__ __align__(16) u16 Ps[2][BM * LDSPITCH];

  const int tid    = threadIdx.x;
  const int lane   = tid & 31;
  const int wave   = tid >> 5;
  const int laneHi = lane >> 4;     // 0/1 selects K sub-slice per WMMA A/B layout
  const int lane16 = lane & 15;

  const int n0 = blockIdx.y * BN;   // x rows base
  const int m0 = blockIdx.x * BM;   // pattern rows base

  const int waveRow = wave & 3;     // 4 slabs of 32 rows
  const int waveCol = wave >> 2;    // 2 slabs of 64 cols

  v8f acc[2][4];
  #pragma unroll
  for (int i = 0; i < 2; ++i)
    #pragma unroll
    for (int j = 0; j < 4; ++j) acc[i][j] = v8f{};

#ifdef HAVE_TDM
  // prime buffer 0
  if (wave == 0) {
    tdm_load_tile((u32)(uintptr_t)&Xs[0][0], xb + (size_t)n0 * D_DIM, (u32)N_ROWS);
    tdm_load_tile((u32)(uintptr_t)&Ps[0][0], pb + (size_t)m0 * D_DIM, (u32)M_ROWS);
  }
#endif

  for (int ci = 0; ci < NCHUNK; ++ci) {
    const int cur = ci & 1;
#ifdef HAVE_TDM
    if (wave == 0) wait_tensor0();      // current buffer's DMA complete
    __syncthreads();                    // release all waves; prev compute done
    if (wave == 0 && ci + 1 < NCHUNK) { // prefetch next chunk into other buffer
      const int kc = (ci + 1) * BK;
      tdm_load_tile((u32)(uintptr_t)&Xs[cur ^ 1][0],
                    xb + (size_t)n0 * D_DIM + kc, (u32)N_ROWS);
      tdm_load_tile((u32)(uintptr_t)&Ps[cur ^ 1][0],
                    pb + (size_t)m0 * D_DIM + kc, (u32)M_ROWS);
    }
#else
    __syncthreads();
    {
      const int kc = ci * BK;
      for (int idx = tid; idx < BN * (BK / 8); idx += 256) {
        int row = idx >> 3, c8 = (idx & 7) * 8;
        *(u16x8*)&Xs[cur][row * LDSPITCH + c8] =
            *(const u16x8*)&xb[(size_t)(n0 + row) * D_DIM + kc + c8];
        *(u16x8*)&Ps[cur][row * LDSPITCH + c8] =
            *(const u16x8*)&pb[(size_t)(m0 + row) * D_DIM + kc + c8];
      }
    }
    __syncthreads();
#endif

    const u16* Xbuf = &Xs[cur][0];
    const u16* Pbuf = &Ps[cur][0];

    #pragma unroll
    for (int ks = 0; ks < BK; ks += 32) {
      const int kOff = ks + laneHi * 8;
      v16bf a[2], b[4];
      #pragma unroll
      for (int sr = 0; sr < 2; ++sr) {
        int row = waveRow * 32 + sr * 16 + lane16;
        const u16* p = &Xbuf[row * LDSPITCH + kOff];
        union { u16x8 h[2]; v16bf v; } u;
        u.h[0] = *(const u16x8*)p;          // K = kOff .. kOff+7
        u.h[1] = *(const u16x8*)(p + 16);   // K = kOff+16 .. kOff+23
        a[sr] = u.v;
      }
      #pragma unroll
      for (int sc = 0; sc < 4; ++sc) {
        int col = waveCol * 64 + sc * 16 + lane16;
        const u16* p = &Pbuf[col * LDSPITCH + kOff];
        union { u16x8 h[2]; v16bf v; } u;
        u.h[0] = *(const u16x8*)p;
        u.h[1] = *(const u16x8*)(p + 16);
        b[sc] = u.v;
      }
      #pragma unroll
      for (int sr = 0; sr < 2; ++sr)
        #pragma unroll
        for (int sc = 0; sc < 4; ++sc)
          acc[sr][sc] = __builtin_amdgcn_wmma_f32_16x16x32_bf16(
              false, a[sr], false, b[sc], (short)0, acc[sr][sc], false, false);
    }
  }

  // ---- fused epilogue: sq-dist -> exp -> masked weighted sums -> atomics ----
  float p2v[4], w0v[4], w1v[4];
  #pragma unroll
  for (int sc = 0; sc < 4; ++sc) {
    int gm = m0 + waveCol * 64 + sc * 16 + lane16;
    p2v[sc] = p2[gm];
    w0v[sc] = w0[gm];
    w1v[sc] = w1[gm];
  }
  #pragma unroll
  for (int sr = 0; sr < 2; ++sr) {
    #pragma unroll
    for (int j = 0; j < 8; ++j) {
      int gn = n0 + waveRow * 32 + sr * 16 + laneHi * 8 + j;  // C layout: VGPR j -> M row
      float x2v = x2[gn];
      float s0 = 0.0f, s1 = 0.0f;
      #pragma unroll
      for (int sc = 0; sc < 4; ++sc) {
        float cross = acc[sr][sc][j];
        float sq = fmaxf(x2v + p2v[sc] - 2.0f * cross, 0.0f);
        float kv = __expf(-0.5f * sq);
        s0 += kv * w0v[sc];
        s1 += kv * w1v[sc];
      }
      #pragma unroll
      for (int o = 1; o <= 8; o <<= 1) {   // reduce within each 16-lane half
        s0 += __shfl_xor(s0, o, 32);
        s1 += __shfl_xor(s1, o, 32);
      }
      if (lane16 == 0) {
        atomicAdd(&out[(size_t)gn * 2 + 0], s0);
        atomicAdd(&out[(size_t)gn * 2 + 1], s1);
      }
    }
  }
}

// ---------- launch ----------

extern "C" void kernel_launch(void* const* d_in, const int* in_sizes, int n_in,
                              void* d_out, int out_size, void* d_ws, size_t ws_size,
                              hipStream_t stream) {
  const float* x       = (const float*)d_in[0];
  const float* pat     = (const float*)d_in[1];
  const int*   targets = (const int*)d_in[2];
  float*       out     = (float*)d_out;

  char* ws = (char*)d_ws;
  size_t off = 0;
  u16* xb = (u16*)(ws + off);  off += (size_t)N_ROWS * D_DIM * sizeof(u16);
  u16* pb = (u16*)(ws + off);  off += (size_t)M_ROWS * D_DIM * sizeof(u16);
  float* x2 = (float*)(ws + off); off += (size_t)N_ROWS * sizeof(float);
  float* p2 = (float*)(ws + off); off += (size_t)M_ROWS * sizeof(float);
  float* w0 = (float*)(ws + off); off += (size_t)M_ROWS * sizeof(float);
  float* w1 = (float*)(ws + off); off += (size_t)M_ROWS * sizeof(float);

  pnn_zero<<<(N_ROWS * 2 + 255) / 256, 256, 0, stream>>>(out, N_ROWS * 2);
  pnn_weights<<<1, 256, 0, stream>>>(targets, w0, w1, M_ROWS);
  pnn_convert<<<N_ROWS / 8, 256, 0, stream>>>(x, xb, x2);
  pnn_convert<<<M_ROWS / 8, 256, 0, stream>>>(pat, pb, p2);

  dim3 grid(M_ROWS / BM, N_ROWS / BN);
  pnn_gemm<<<grid, 256, 0, stream>>>(xb, pb, x2, p2, w0, w1, out);
}
